// CodeGNN_61031485276239
// MI455X (gfx1250) — compile-verified, matrix operand order checked by reading, not run on hardware
//
#include <hip/hip_runtime.h>
#include <hip/hip_bf16.h>

// Problem constants (match reference).
#define N_    50000
#define E_    500000
#define F_    128
#define H_    128
#define OUT_  768
#define MAXKSTEPS 12   // Kdim up to 384 -> 12 steps of 32

typedef __attribute__((ext_vector_type(16))) __bf16 v16bf;
typedef __attribute__((ext_vector_type(8)))  float  v8f;
typedef int v4i_ __attribute__((vector_size(16)));

#if __has_builtin(__builtin_amdgcn_global_load_async_to_lds_b128)
#define ASYNC_LDS 1
typedef __attribute__((address_space(1))) v4i_* as1_v4i;
typedef __attribute__((address_space(3))) v4i_* as3_v4i;
#endif

// ---------- small helpers ----------
__device__ __forceinline__ unsigned short f2bf(float f) {
  unsigned u = __float_as_uint(f);
  unsigned r = ((u >> 16) & 1u) + 0x7fffu;   // round-to-nearest-even
  return (unsigned short)((u + r) >> 16);
}
// monotonic float<->uint mapping for atomic max on floats
__device__ __forceinline__ unsigned f2o(float f) {
  unsigned u = __float_as_uint(f);
  return (u & 0x80000000u) ? ~u : (u | 0x80000000u);
}
__device__ __forceinline__ float o2f(unsigned o) {
  unsigned u = (o & 0x80000000u) ? (o & 0x7fffffffu) : ~o;
  return __uint_as_float(u);
}
#define NEG_INF_ORD 0x007FFFFFu   // f2o(-inf)

// 16-byte chunk copy global->LDS (async-to-LDS path when available)
__device__ __forceinline__ void lds_copy16(const void* g, void* l) {
#ifdef ASYNC_LDS
  __builtin_amdgcn_global_load_async_to_lds_b128(
      (as1_v4i)(v4i_*)g, (as3_v4i)(v4i_*)l, 0, 0);
#else
  *(uint4*)l = *(const uint4*)g;
#endif
}
__device__ __forceinline__ void async_join() {
#ifdef ASYNC_LDS
#if __has_builtin(__builtin_amdgcn_s_wait_asynccnt)
  __builtin_amdgcn_s_wait_asynccnt(0);
#else
  asm volatile("s_wait_asynccnt 0x0" ::: "memory");
#endif
#endif
  __syncthreads();
}

// ---------- fills ----------
__global__ void fill_u32(unsigned* p, long long n, unsigned v) {
  long long i = (long long)blockIdx.x * blockDim.x + threadIdx.x;
  long long stride = (long long)gridDim.x * blockDim.x;
  for (; i < n; i += stride) p[i] = v;
}
__global__ void fill_f32(float* p, long long n, float v) {
  long long i = (long long)blockIdx.x * blockDim.x + threadIdx.x;
  long long stride = (long long)gridDim.x * blockDim.x;
  for (; i < n; i += stride) p[i] = v;
}

// ---------- fp32 -> bf16 conversion ----------
__global__ void cvt_bf16(const float* __restrict__ s, unsigned short* __restrict__ d,
                         long long n) {
  long long i = (long long)blockIdx.x * blockDim.x + threadIdx.x;
  long long stride = (long long)gridDim.x * blockDim.x;
  for (; i < n; i += stride) d[i] = f2bf(s[i]);
}

// ---------- weight packing: fp32 [Kdim,Ncols] -> bf16 WMMA fragments ----------
// layout: [kstep][ntile][lane][16 bf16]  (32 contiguous bytes per lane)
__device__ __forceinline__ int k_in_tile(int lane, int e) {
  return ((e < 8) ? e : e + 8) + ((lane >= 16) ? 8 : 0);
}
__global__ void pack_w(const float* __restrict__ W, unsigned short* __restrict__ out,
                       int Kdim, int Ncols) {
  int t = blockIdx.x * blockDim.x + threadIdx.x;
  int ntiles = Ncols / 16, ksteps = Kdim / 32;
  if (t >= ksteps * ntiles * 32) return;
  int lane = t & 31;
  int rest = t >> 5;
  int ntile = rest % ntiles;
  int kstep = rest / ntiles;
  int n = ntile * 16 + (lane & 15);
  unsigned short* dstp = out + (size_t)t * 16;
#pragma unroll
  for (int e = 0; e < 16; e++) {
    int k = kstep * 32 + k_in_tile(lane, e);
    dstp[e] = f2bf(W[(size_t)k * Ncols + n]);
  }
}

// A-tile chunk decomposition: chunk c (of ksteps*64) is 8 consecutive K values
// of one row, landing at a contiguous 16B LDS slot in fragment order.
__device__ __forceinline__ void chunk_decode(int c, int rowBase,
                                             int& kstep, int& row, int& k0,
                                             unsigned& ldsOff) {
  kstep = c >> 6;
  int r = c & 63, lane2 = r >> 1, half = r & 1;
  row = rowBase + (lane2 & 15);
  k0 = kstep * 32 + ((lane2 >= 16) ? 8 : 0) + (half ? 16 : 0);
  ldsOff = (unsigned)((((kstep * 32 + lane2) * 16) + half * 8) * 2);
}

// ---------- generic node GEMM (all-bf16 A segments, no gather) ----------
// block = 256 (8 waves). gridDim.x = Rows/16, gridDim.y = Ncols/128.
__global__ __launch_bounds__(256) void gemm_node(
    const unsigned short* __restrict__ p0, int l0,
    const unsigned short* __restrict__ p1, int l1,
    const unsigned short* __restrict__ p2, int l2,
    const unsigned short* __restrict__ Bp, const float* __restrict__ bias,
    float* __restrict__ out, int Ncols, int Kdim) {
  __shared__ alignas(32) unsigned short As[MAXKSTEPS * 512];
  const int ksteps  = Kdim / 32;
  const int ntilesB = Ncols / 16;
  const int rowBase = blockIdx.x * 16;
  const int tid = threadIdx.x;
  for (int c = tid; c < ksteps * 64; c += 256) {
    int kstep, row, k0; unsigned ldsOff;
    chunk_decode(c, rowBase, kstep, row, k0, ldsOff);
    const unsigned short* p; int sb, sl;
    if (k0 < l0)           { p = p0; sb = 0;       sl = l0; }
    else if (k0 < l0 + l1) { p = p1; sb = l0;      sl = l1; }
    else                   { p = p2; sb = l0 + l1; sl = l2; }
    lds_copy16(p + (size_t)row * sl + (k0 - sb), (char*)As + ldsOff);
  }
  async_join();
  const int lane = tid & 31, wave = tid >> 5;
  const int ntile = blockIdx.y * 8 + wave;
  v8f acc = {};
  const v16bf* Afr = (const v16bf*)As;
  const v16bf* Bfr = (const v16bf*)Bp;
  for (int ks = 0; ks < ksteps; ks++) {
    v16bf a = Afr[ks * 32 + lane];
    v16bf b = Bfr[((size_t)ks * ntilesB + ntile) * 32 + lane];
    acc = __builtin_amdgcn_wmma_f32_16x16x32_bf16(false, a, false, b,
                                                  (short)0, acc, false, false);
  }
  const int n  = ntile * 16 + (lane & 15);
  const int mo = (lane >= 16) ? 8 : 0;
  const float bs = bias ? bias[n] : 0.f;
#pragma unroll
  for (int r = 0; r < 8; r++) {
    int row = rowBase + r + mo;
    out[(size_t)row * Ncols + n] = acc[r] + bs;
  }
}

// ---------- edge GEMM: eK & eV; epilogue fuses logit + segment atomic-max ----
// A row = [ kind_bf[src[e]] | cvt(edge_h[e]) | (h_bf[src[e]]) ], Kdim 256/384.
__global__ __launch_bounds__(256) void gemm_edge(
    const unsigned short* __restrict__ kind_bf, const float* __restrict__ edge_h,
    const unsigned short* __restrict__ h_bf,
    const int* __restrict__ src, const int* __restrict__ dst,
    const unsigned short* __restrict__ Kp, const unsigned short* __restrict__ Vp,
    const float* __restrict__ Kb, const float* __restrict__ Vb,
    const float* __restrict__ Qbuf,
    float* __restrict__ logit, float* __restrict__ eVout,
    unsigned* __restrict__ amaxo, int Kdim) {
  __shared__ alignas(32) unsigned short As[MAXKSTEPS * 512];
  const int ksteps = Kdim / 32;
  const int eBase  = blockIdx.x * 16;
  const int tid = threadIdx.x;
  for (int c = tid; c < ksteps * 64; c += 256) {
    int kstep, row, k0; unsigned ldsOff;
    chunk_decode(c, eBase, kstep, row, k0, ldsOff);
    if (kstep < 4) {                 // kind[src[e]] : bf16, gathered (L2-hot)
      lds_copy16(kind_bf + (size_t)src[row] * F_ + k0, (char*)As + ldsOff);
    } else if (kstep < 8) {          // edge_h : fp32 streamed once, cvt here
      const float* gp = edge_h + (size_t)row * F_ + (k0 - F_);
      float4 a = *(const float4*)gp;
      float4 b = *(const float4*)(gp + 4);
      uint4 v;
      v.x = (unsigned)f2bf(a.x) | ((unsigned)f2bf(a.y) << 16);
      v.y = (unsigned)f2bf(a.z) | ((unsigned)f2bf(a.w) << 16);
      v.z = (unsigned)f2bf(b.x) | ((unsigned)f2bf(b.y) << 16);
      v.w = (unsigned)f2bf(b.z) | ((unsigned)f2bf(b.w) << 16);
      *(uint4*)((char*)As + ldsOff) = v;
    } else {                         // h[src[e]] : bf16, gathered (layer 2)
      lds_copy16(h_bf + (size_t)src[row] * H_ + (k0 - 2 * F_), (char*)As + ldsOff);
    }
  }
  async_join();
  const int lane = tid & 31, wave = tid >> 5;    // 8 waves -> H=128 columns
  v8f cK = {}, cV = {};
  const v16bf* Afr = (const v16bf*)As;
  const v16bf* Kf  = (const v16bf*)Kp;
  const v16bf* Vf  = (const v16bf*)Vp;
  for (int ks = 0; ks < ksteps; ks++) {
    v16bf a  = Afr[ks * 32 + lane];
    v16bf bk = Kf[((size_t)ks * 8 + wave) * 32 + lane];
    v16bf bv = Vf[((size_t)ks * 8 + wave) * 32 + lane];
    cK = __builtin_amdgcn_wmma_f32_16x16x32_bf16(false, a, false, bk, (short)0, cK, false, false);
    cV = __builtin_amdgcn_wmma_f32_16x16x32_bf16(false, a, false, bv, (short)0, cV, false, false);
  }
  const int n  = wave * 16 + (lane & 15);
  const int mo = (lane >= 16) ? 8 : 0;
  const float kb = Kb[n], vb = Vb[n];
#pragma unroll
  for (int r = 0; r < 8; r++) {
    int e = eBase + r + mo;
    int d = dst[e];
    size_t dn = (size_t)d * H_ + n;
    float lg = (cK[r] + kb) * Qbuf[dn];
    logit[(size_t)e * H_ + n] = lg;
    atomicMax(&amaxo[dn], f2o(lg));          // fused segment max
    eVout[(size_t)e * H_ + n] = cV[r] + vb;
  }
}

// ---------- in-place row LayerNorm ----------
__global__ void ln_rows(float* __restrict__ x, const float* __restrict__ g,
                        const float* __restrict__ b, int C) {
  __shared__ float red[256];
  const int tid = threadIdx.x, bs = blockDim.x;
  float* xr = x + (size_t)blockIdx.x * C;
  float s = 0.f, ss = 0.f;
  for (int c = tid; c < C; c += bs) { float v = xr[c]; s += v; ss += v * v; }
  red[tid] = s; __syncthreads();
  for (int o = bs >> 1; o > 0; o >>= 1) { if (tid < o) red[tid] += red[tid + o]; __syncthreads(); }
  float mean = red[0] / C;
  __syncthreads();
  red[tid] = ss; __syncthreads();
  for (int o = bs >> 1; o > 0; o >>= 1) { if (tid < o) red[tid] += red[tid + o]; __syncthreads(); }
  float var = red[0] / C - mean * mean;
  float rs = rsqrtf(var + 1e-5f);
  for (int c = tid; c < C; c += bs) xr[c] = (xr[c] - mean) * rs * g[c] + b[c];
}

// ---------- edge softmax passes 2 & 3 (segment over dst, per channel) -------
__global__ void edge_exp(float* __restrict__ logit, const int* __restrict__ dst,
                         const unsigned* __restrict__ amaxo, float* __restrict__ denom) {
  long long i = (long long)blockIdx.x * 256 + threadIdx.x;
  if (i >= (long long)E_ * H_) return;
  int n = (int)(i & (H_ - 1));
  long long e = i >> 7;
  size_t dn = (size_t)dst[e] * H_ + n;
  float ex = expf(logit[i] - o2f(amaxo[dn]));
  logit[i] = ex;
  atomicAdd(&denom[dn], ex);
}
__global__ void edge_agg(const float* __restrict__ ex, const float* __restrict__ eV,
                         const int* __restrict__ dst, const float* __restrict__ denom,
                         float* __restrict__ hn) {
  long long i = (long long)blockIdx.x * 256 + threadIdx.x;
  if (i >= (long long)E_ * H_) return;
  int n = (int)(i & (H_ - 1));
  long long e = i >> 7;
  size_t dn = (size_t)dst[e] * H_ + n;
  float alpha = ex[i] / denom[dn];
  atomicAdd(&hn[dn], alpha * eV[i]);
}

// ---------- global attention pooling ----------
__global__ __launch_bounds__(256) void gate_dot(const float* __restrict__ h1,
    const float* __restrict__ gw, const float* __restrict__ gb, float* __restrict__ g) {
  int wid  = (int)(((long long)blockIdx.x * blockDim.x + threadIdx.x) >> 5);
  int lane = threadIdx.x & 31;
  if (wid >= N_) return;
  const float* row = h1 + (size_t)wid * OUT_;
  float s = 0.f;
  for (int c = lane; c < OUT_; c += 32) s += row[c] * gw[c];
  for (int o = 16; o > 0; o >>= 1) s += __shfl_down(s, o, 32);
  if (lane == 0) g[wid] = s + gb[0];
}
__global__ void gate_max(const float* __restrict__ g, unsigned* __restrict__ gmaxo) {
  __shared__ float red[256];
  int tid = threadIdx.x;
  float m = -INFINITY;
  for (long long i = (long long)blockIdx.x * 256 + tid; i < N_; i += 256LL * gridDim.x)
    m = fmaxf(m, g[i]);
  red[tid] = m; __syncthreads();
  for (int o = 128; o > 0; o >>= 1) { if (tid < o) red[tid] = fmaxf(red[tid], red[tid + o]); __syncthreads(); }
  if (tid == 0) atomicMax(gmaxo, f2o(red[0]));
}
__global__ void gate_sum(const float* __restrict__ g, const unsigned* __restrict__ gmaxo,
                         float* __restrict__ gsum) {
  __shared__ float red[256];
  int tid = threadIdx.x;
  float gm = o2f(*gmaxo);
  float s = 0.f;
  for (long long i = (long long)blockIdx.x * 256 + tid; i < N_; i += 256LL * gridDim.x)
    s += expf(g[i] - gm);
  red[tid] = s; __syncthreads();
  for (int o = 128; o > 0; o >>= 1) { if (tid < o) red[tid] += red[tid + o]; __syncthreads(); }
  if (tid == 0) atomicAdd(gsum, red[0]);
}
__global__ __launch_bounds__(256) void gate_out(const float* __restrict__ h1,
    const float* __restrict__ g, const unsigned* __restrict__ gmaxo,
    const float* __restrict__ gsum, float* __restrict__ out) {
  __shared__ float acc[OUT_];
  int tid = threadIdx.x;
  for (int c = tid; c < OUT_; c += 256) acc[c] = 0.f;
  __syncthreads();
  float gm = o2f(*gmaxo);
  float inv = 1.f / (*gsum);
  int r0 = blockIdx.x * 128;
  int r1 = r0 + 128; if (r1 > N_) r1 = N_;
  for (int r = r0; r < r1; r++) {
    float w = expf(g[r] - gm) * inv;
    const float* row = h1 + (size_t)r * OUT_;
    for (int c = tid; c < OUT_; c += 256) acc[c] += w * row[c];  // c owned by one thread
  }
  for (int c = tid; c < OUT_; c += 256) atomicAdd(&out[c], acc[c]);
}

// ---------- host ----------
extern "C" void kernel_launch(void* const* d_in, const int* in_sizes, int n_in,
                              void* d_out, int out_size, void* d_ws, size_t ws_size,
                              hipStream_t stream) {
  const float* kind   = (const float*)d_in[0];
  const float* edge_h = (const float*)d_in[1];
  const int*   src    = (const int*)  d_in[2];
  const int*   dst    = (const int*)  d_in[3];
  const float* KW  = (const float*)d_in[4];  const float* Kb  = (const float*)d_in[5];
  const float* VW  = (const float*)d_in[6];  const float* Vb  = (const float*)d_in[7];
  const float* QW  = (const float*)d_in[8];  const float* Qb  = (const float*)d_in[9];
  const float* WW  = (const float*)d_in[10]; const float* Wb  = (const float*)d_in[11];
  const float* K2W = (const float*)d_in[12]; const float* K2b = (const float*)d_in[13];
  const float* V2W = (const float*)d_in[14]; const float* V2b = (const float*)d_in[15];
  const float* Q2W = (const float*)d_in[16]; const float* Q2b = (const float*)d_in[17];
  const float* W2W = (const float*)d_in[18]; const float* W2b = (const float*)d_in[19];
  const float* ln1g = (const float*)d_in[20]; const float* ln1b = (const float*)d_in[21];
  const float* ln2g = (const float*)d_in[22]; const float* ln2b = (const float*)d_in[23];
  const float* gw  = (const float*)d_in[24]; const float* gb  = (const float*)d_in[25];
  float* out = (float*)d_out;

  // workspace carve-out (256B aligned)
  char* base = (char*)d_ws;
  size_t off = 0;
  auto carve = [&](size_t bytes) -> char* {
    char* p = base + off;
    off = (off + bytes + 255) & ~(size_t)255;
    return p;
  };
  float*    Qbuf  = (float*)   carve((size_t)N_ * H_ * 4);
  float*    hbuf  = (float*)   carve((size_t)N_ * H_ * 4);
  float*    hnbuf = (float*)   carve((size_t)N_ * H_ * 4);
  unsigned* amaxo = (unsigned*)carve((size_t)N_ * H_ * 4);
  float*    denom = (float*)   carve((size_t)N_ * H_ * 4);
  float*    h1buf = (float*)   carve((size_t)N_ * OUT_ * 4);
  float*    logit = (float*)   carve((size_t)E_ * H_ * 4);
  float*    eVbuf = (float*)   carve((size_t)E_ * H_ * 4);
  float*    gbuf  = (float*)   carve((size_t)N_ * 4);
  unsigned* gmaxo = (unsigned*)carve(4);
  float*    gsum  = (float*)   carve(4);
  unsigned short* kind_bf = (unsigned short*)carve((size_t)N_ * F_ * 2);
  unsigned short* h_bf    = (unsigned short*)carve((size_t)N_ * H_ * 2);
  unsigned short* hn_bf   = (unsigned short*)carve((size_t)N_ * H_ * 2);
  unsigned short* KWp  = (unsigned short*)carve((size_t)256 * 128 * 2);
  unsigned short* VWp  = (unsigned short*)carve((size_t)256 * 128 * 2);
  unsigned short* QWp  = (unsigned short*)carve((size_t)128 * 128 * 2);
  unsigned short* WWp  = (unsigned short*)carve((size_t)256 * 128 * 2);
  unsigned short* K2Wp = (unsigned short*)carve((size_t)384 * 128 * 2);
  unsigned short* V2Wp = (unsigned short*)carve((size_t)384 * 128 * 2);
  unsigned short* Q2Wp = (unsigned short*)carve((size_t)256 * 128 * 2);
  unsigned short* W2Wp = (unsigned short*)carve((size_t)384 * 768 * 2);
  (void)ws_size; (void)in_sizes; (void)n_in; (void)out_size;

  const long long NH = (long long)N_ * H_;
  const long long EHblocks = ((long long)E_ * H_ + 255) / 256;

  auto pack = [&](const float* W, int Kd, int Nc, unsigned short* p) {
    int total = (Kd / 32) * (Nc / 16) * 32;
    pack_w<<<(total + 255) / 256, 256, 0, stream>>>(W, p, Kd, Nc);
  };
  pack(KW, 256, 128, KWp);   pack(VW, 256, 128, VWp);
  pack(QW, 128, 128, QWp);   pack(WW, 256, 128, WWp);
  pack(K2W, 384, 128, K2Wp); pack(V2W, 384, 128, V2Wp);
  pack(Q2W, 256, 128, Q2Wp); pack(W2W, 384, 768, W2Wp);

  // init accumulators + bf16 copy of kind
  fill_u32<<<4096, 256, 0, stream>>>(amaxo, NH, NEG_INF_ORD);
  fill_f32<<<4096, 256, 0, stream>>>(denom, NH, 0.f);
  fill_f32<<<4096, 256, 0, stream>>>(hnbuf, NH, 0.f);
  fill_u32<<<1, 32, 0, stream>>>(gmaxo, 1, NEG_INF_ORD);
  fill_f32<<<1, 32, 0, stream>>>(gsum, 1, 0.f);
  fill_f32<<<3, 256, 0, stream>>>(out, OUT_, 0.f);
  cvt_bf16<<<4096, 256, 0, stream>>>(kind, kind_bf, (long long)N_ * F_);

  // ---- layer 1 ----
  gemm_node<<<dim3(N_ / 16, 1), 256, 0, stream>>>(
      kind_bf, F_, nullptr, 0, nullptr, 0, QWp, Qb, Qbuf, H_, 128);
  gemm_edge<<<E_ / 16, 256, 0, stream>>>(
      kind_bf, edge_h, nullptr, src, dst, KWp, VWp, Kb, Vb, Qbuf,
      logit, eVbuf, amaxo, 256);
  edge_exp<<<EHblocks, 256, 0, stream>>>(logit, dst, amaxo, denom);
  edge_agg<<<EHblocks, 256, 0, stream>>>(logit, eVbuf, dst, denom, hnbuf);
  cvt_bf16<<<4096, 256, 0, stream>>>(hnbuf, hn_bf, NH);
  gemm_node<<<dim3(N_ / 16, 1), 256, 0, stream>>>(
      hn_bf, H_, kind_bf, F_, nullptr, 0, WWp, Wb, hbuf, H_, 256);
  ln_rows<<<N_, 128, 0, stream>>>(hbuf, ln1g, ln1b, H_);
  cvt_bf16<<<4096, 256, 0, stream>>>(hbuf, h_bf, NH);

  // reset accumulators for layer 2
  fill_u32<<<4096, 256, 0, stream>>>(amaxo, NH, NEG_INF_ORD);
  fill_f32<<<4096, 256, 0, stream>>>(denom, NH, 0.f);
  fill_f32<<<4096, 256, 0, stream>>>(hnbuf, NH, 0.f);

  // ---- layer 2 ----
  gemm_node<<<dim3(N_ / 16, 1), 256, 0, stream>>>(
      kind_bf, F_, h_bf, H_, nullptr, 0, Q2Wp, Q2b, Qbuf, H_, 256);
  gemm_edge<<<E_ / 16, 256, 0, stream>>>(
      kind_bf, edge_h, h_bf, src, dst, K2Wp, V2Wp, K2b, V2b, Qbuf,
      logit, eVbuf, amaxo, 384);
  edge_exp<<<EHblocks, 256, 0, stream>>>(logit, dst, amaxo, denom);
  edge_agg<<<EHblocks, 256, 0, stream>>>(logit, eVbuf, dst, denom, hnbuf);
  cvt_bf16<<<4096, 256, 0, stream>>>(hnbuf, hn_bf, NH);
  gemm_node<<<dim3(N_ / 16, OUT_ / 128), 256, 0, stream>>>(
      hn_bf, H_, h_bf, H_, kind_bf, F_, W2Wp, W2b, h1buf, OUT_, 384);
  ln_rows<<<N_, 256, 0, stream>>>(h1buf, ln2g, ln2b, OUT_);

  // ---- global attention pooling ----
  gate_dot<<<(N_ * 32 + 255) / 256, 256, 0, stream>>>(h1buf, gw, gb, gbuf);
  gate_max<<<256, 256, 0, stream>>>(gbuf, gmaxo);
  gate_sum<<<256, 256, 0, stream>>>(gbuf, gmaxo, gsum);
  gate_out<<<(N_ + 127) / 128, 256, 0, stream>>>(h1buf, gbuf, gmaxo, gsum, out);
}
